// DeltaRetroModel_3204045603545
// MI455X (gfx1250) — matrix operand order repaired and checked by reading, not compiled, not run on hardware
//
#include <hip/hip_runtime.h>
#include <hip/hip_bf16.h>
#include <math.h>

typedef __attribute__((ext_vector_type(16))) _Float16 v16h;
typedef __attribute__((ext_vector_type(8)))  float    v8f;

#define H   64
#define H2  128
#define VOC 64
#define BSZ 256
#define LSQ 2048

// ---------------------------------------------------------------------------
// WMMA fragment gathers (CDNA5 16x16x32 f16 layouts, wave32)
// A (16x32, MxK): lanes 0-15 -> M=lane, K = {0..7,16..23}+kbase
//                 lanes 16-31 -> M=lane-16, K = {8..15,24..31}+kbase
// B (32x16, KxN): lanes 0-15 -> N=lane,   K = {0..7,16..23}+kbase
//                 lanes 16-31 -> N=lane-16, K = {8..15,24..31}+kbase
// C/D (16x16 f32): VGPR r -> (M=r, N=lane) lanes<16 ; (M=r+8, N=lane-16)
// ---------------------------------------------------------------------------
__device__ inline v16h load_A_f16(const _Float16* tile, int ld, int kbase, int lane) {
    v16h a;
    const int row  = lane & 15;
    const int koff = (lane & 16) ? 8 : 0;
    const _Float16* p = tile + row * ld + kbase + koff;
#pragma unroll
    for (int h = 0; h < 8; ++h) a[h] = p[h];
#pragma unroll
    for (int h = 0; h < 8; ++h) a[8 + h] = p[16 + h];
    return a;
}

// B[k][n] = W[nbase+n][kbase+k]  (i.e. B operand is W^T; W stored row-major [N][K])
__device__ inline v16h load_B_WT(const _Float16* w, int ld, int nbase, int kbase, int lane) {
    v16h bf;
    const int n  = lane & 15;
    const int k0 = (lane & 16) ? 8 : 0;
    const _Float16* p = w + (nbase + n) * ld + kbase + k0;
#pragma unroll
    for (int j = 0; j < 8; ++j) bf[j] = p[j];
#pragma unroll
    for (int j = 0; j < 8; ++j) bf[8 + j] = p[16 + j];
    return bf;
}

// ---------------------------------------------------------------------------
// Kernel 1: token encoder.  h = LN(e + relu(e@W1^T + b1)@W2^T + b2)
// 2 waves / WG, 16 tokens per wave.  grid = B*L/32
// ---------------------------------------------------------------------------
__global__ __launch_bounds__(64) void enc_kernel(
    const int* __restrict__ x, const float* __restrict__ embed,
    const float* __restrict__ W1, const float* __restrict__ b1,
    const float* __restrict__ W2, const float* __restrict__ b2,
    const float* __restrict__ ln_g, const float* __restrict__ ln_b,
    float* __restrict__ h_enc)
{
    __shared__ _Float16 w1h[H2 * H];       // 16 KB  W1 [128][64] f16
    __shared__ _Float16 w2h[H * H2];       // 16 KB  W2 [64][128] f16
    __shared__ _Float16 hin_h[2][16 * H];  //  4 KB  embedding tile f16
    __shared__ _Float16 ffh[2][16 * H2];   //  8 KB  relu(GEMM1) f16
    __shared__ float    outf[2][16 * 68];  // ~8.7KB GEMM2 + residual (padded)
    __shared__ float    mus[2][16], rsg[2][16];

    const int tid  = threadIdx.x;
    const int wave = tid >> 5;
    const int lane = tid & 31;

    // cooperative weight convert f32 -> f16 in LDS
    for (int i = tid; i < H2 * H; i += 64) w1h[i] = (_Float16)W1[i];
    for (int i = tid; i < H * H2; i += 64) w2h[i] = (_Float16)W2[i];
    __syncthreads();

    const long tokBase = ((long)blockIdx.x * 2 + wave) * 16;

    // gather 16 token embeddings (1024 f32) into f16 staging
    for (int e = lane; e < 16 * H; e += 32) {
        const int r = e >> 6, c = e & 63;
        const int tok = x[tokBase + r];
        hin_h[wave][e] = (_Float16)embed[tok * H + c];
    }
    __syncthreads();

    const int col16 = lane & 15;
    const int rbase = (lane & 16) ? 8 : 0;

    // GEMM1: [16x64] @ W1^T -> [16x128], +b1, relu
#pragma unroll
    for (int nt = 0; nt < 8; ++nt) {
        v8f acc = {};
#pragma unroll
        for (int kt = 0; kt < 2; ++kt) {
            v16h a = load_A_f16(&hin_h[wave][0], H, kt * 32, lane);
            v16h bm = load_B_WT(w1h, H, nt * 16, kt * 32, lane);
            acc = __builtin_amdgcn_wmma_f32_16x16x32_f16(
                false, a, false, bm, (short)0, acc, false, false);
        }
        const int col = nt * 16 + col16;
        const float bias = b1[col];
#pragma unroll
        for (int r = 0; r < 8; ++r) {
            float v = acc[r] + bias;
            ffh[wave][(rbase + r) * H2 + col] = (_Float16)fmaxf(v, 0.f);
        }
    }
    __syncthreads();

    // GEMM2: [16x128] @ W2^T -> [16x64], +b2 + residual
#pragma unroll
    for (int nt = 0; nt < 4; ++nt) {
        v8f acc = {};
#pragma unroll
        for (int kt = 0; kt < 4; ++kt) {
            v16h a = load_A_f16(&ffh[wave][0], H2, kt * 32, lane);
            v16h bm = load_B_WT(w2h, H2, nt * 16, kt * 32, lane);
            acc = __builtin_amdgcn_wmma_f32_16x16x32_f16(
                false, a, false, bm, (short)0, acc, false, false);
        }
        const int col = nt * 16 + col16;
        const float bias = b2[col];
#pragma unroll
        for (int r = 0; r < 8; ++r) {
            const int row = rbase + r;
            outf[wave][row * 68 + col] =
                acc[r] + bias + (float)hin_h[wave][row * H + col];
        }
    }
    __syncthreads();

    // LayerNorm stats, one row per lane (lanes 0..15)
    if (lane < 16) {
        float s = 0.f, s2 = 0.f;
        for (int c = 0; c < H; ++c) {
            float v = outf[wave][lane * 68 + c];
            s += v; s2 += v * v;
        }
        const float mu = s * (1.f / H);
        mus[wave][lane] = mu;
        rsg[wave][lane] = rsqrtf(s2 * (1.f / H) - mu * mu + 1e-5f);
    }
    __syncthreads();

    // normalized, coalesced store
    for (int e = lane; e < 16 * H; e += 32) {
        const int r = e >> 6, c = e & 63;
        float v = (outf[wave][r * 68 + c] - mus[wave][r]) * rsg[wave][r]
                    * ln_g[c] + ln_b[c];
        h_enc[(tokBase + r) * H + c] = v;
    }
}

// ---------------------------------------------------------------------------
// Kernel 2: gated delta-rule scan + retrieval readout.
// One wave32 per batch element; lane owns rows (tid, tid+32) of M in VGPRs.
// __syncthreads() in single-wave WGs lowers to S_NOP (+ counter waits).
// h_t fetch is software-pipelined one step ahead: the global load for t+1
// issues before step t's matvecs; its loadcnt wait lands at the NEXT
// iteration's LDS store, off the serial dependence chain.
// ---------------------------------------------------------------------------
__device__ inline float waveSum(float v) {
#pragma unroll
    for (int off = 16; off > 0; off >>= 1)
        v += __shfl_xor(v, off, 32);
    return v;
}

__global__ __launch_bounds__(32) void scan_kernel(
    const float* __restrict__ h_enc,
    const float* __restrict__ Wk, const float* __restrict__ Wv,
    const float* __restrict__ Wq, const float* __restrict__ Wr,
    const float* __restrict__ alpha_p,
    const float* __restrict__ Wout, const float* __restrict__ bout,
    float* __restrict__ out)
{
    __shared__ float WkL[H * 65];   // padded: bank = (row+col) % 64, conflict-free
    __shared__ float WvL[H * 65];
    __shared__ float ML [H * 65];
    __shared__ float ht[H], kv[H], qv[H], qr[H], cn[H], mixedL[H];
    __shared__ float attL[8];
    __shared__ int   topi[8];

    const int b   = blockIdx.x;
    const int tid = threadIdx.x;         // 0..31
    const int j1 = tid, j2 = tid + 32;

    for (int e = tid; e < H * H; e += 32) {
        const int r = e >> 6, c = e & 63;
        WkL[r * 65 + c] = Wk[e];
        WvL[r * 65 + c] = Wv[e];
    }
    float M1[H], M2[H];
#pragma unroll
    for (int m = 0; m < H; ++m) { M1[m] = 0.f; M2[m] = 0.f; }

    const float* hb = h_enc + (size_t)b * LSQ * H;

    // pipeline prologue: fetch h_0 into registers
    float h1n = hb[j1];
    float h2n = hb[j2];
    __syncthreads();

    for (int t = 0; t < LSQ; ++t) {
        // publish current h_t (loaded last iteration), issue load for t+1
        ht[j1] = h1n;
        ht[j2] = h2n;
        if (t + 1 < LSQ) {
            h1n = hb[(t + 1) * H + j1];
            h2n = hb[(t + 1) * H + j2];
        }
        __syncthreads();

        float k1 = 0.f, k2 = 0.f, v1 = 0.f, v2 = 0.f;
#pragma unroll
        for (int m = 0; m < H; ++m) {
            const float hm = ht[m];
            k1 += WkL[j1 * 65 + m] * hm;
            k2 += WkL[j2 * 65 + m] * hm;
            v1 += WvL[j1 * 65 + m] * hm;
            v2 += WvL[j2 * 65 + m] * hm;
        }
        const float kk  = waveSum(k1 * k1 + k2 * k2);
        const float inv = 1.f / fmaxf(sqrtf(kk), 1e-12f);
        k1 *= inv; k2 *= inv;
        kv[j1] = k1; kv[j2] = k2;
        __syncthreads();

        float p1 = 0.f, p2 = 0.f;
#pragma unroll
        for (int m = 0; m < H; ++m) {
            const float km = kv[m];
            p1 += M1[m] * km;
            p2 += M2[m] * km;
        }
        const float d1 = v1 - p1, d2 = v2 - p2;
        const float nd2 = waveSum(d1 * d1 + d2 * d2);
        const float nv2 = waveSum(v1 * v1 + v2 * v2);
        if (nd2 > 0.16f * nv2) {        // ||d|| > 0.4*||v||  (uniform branch)
#pragma unroll
            for (int m = 0; m < H; ++m) {
                const float km = kv[m];
                M1[m] += d1 * km;
                M2[m] += d2 * km;
            }
        }
        __syncthreads();                // fence ht/kv before next overwrite
    }

    // ---- readout: M -> LDS (padded), ht still holds h[b, L-1, :] ----
#pragma unroll
    for (int m = 0; m < H; ++m) {
        ML[j1 * 65 + m] = M1[m];
        ML[j2 * 65 + m] = M2[m];
    }
    __syncthreads();

    float q1 = 0.f, q2 = 0.f;
    for (int m = 0; m < H; ++m) {
        const float hm = ht[m];
        q1 += Wq[j1 * H + m] * hm;
        q2 += Wq[j2 * H + m] * hm;
    }
    qv[j1] = q1; qv[j2] = q2;
    __syncthreads();

    float r1 = 0.f, r2 = 0.f;
    for (int m = 0; m < H; ++m) {
        const float qm = qv[m];
        r1 += Wr[j1 * H + m] * qm;
        r2 += Wr[j2 * H + m] * qm;
    }
    qr[j1] = r1; qr[j2] = r2;

    // slot norms = column norms of M (conflict-free column walk)
    float c1 = 0.f, c2 = 0.f;
    for (int h = 0; h < H; ++h) {
        const float a1 = ML[h * 65 + j1]; c1 += a1 * a1;
        const float a2 = ML[h * 65 + j2]; c2 += a2 * a2;
    }
    cn[j1] = sqrtf(c1); cn[j2] = sqrtf(c2);
    __syncthreads();

    if (tid == 0) {                      // top-8, ties -> lowest index
        for (int k = 0; k < 8; ++k) {
            int best = 0; float bv = cn[0];
            for (int i = 1; i < H; ++i)
                if (cn[i] > bv) { bv = cn[i]; best = i; }
            topi[k] = best; cn[best] = -1.f;
        }
    }
    __syncthreads();

    if (tid < 8) {                       // logits = <sel_k, q_r> / sqrt(H)
        const int ci = topi[tid];
        float lg = 0.f;
        for (int h = 0; h < H; ++h) lg += ML[h * 65 + ci] * qr[h];
        attL[tid] = lg * 0.125f;
    }
    __syncthreads();

    // softmax (redundant per lane), retro ctx + memory ctx
    float mx = attL[0];
#pragma unroll
    for (int k = 1; k < 8; ++k) mx = fmaxf(mx, attL[k]);
    float w8[8], ssum = 0.f;
#pragma unroll
    for (int k = 0; k < 8; ++k) { w8[k] = __expf(attL[k] - mx); ssum += w8[k]; }
    const float invs = 1.f / ssum;

    float re1 = 0.f, re2 = 0.f;
#pragma unroll
    for (int k = 0; k < 8; ++k) {
        const int ci = topi[k];
        const float wk_ = w8[k] * invs;
        re1 += wk_ * ML[j1 * 65 + ci];
        re2 += wk_ * ML[j2 * 65 + ci];
    }
    float mc1 = 0.f, mc2 = 0.f;
#pragma unroll
    for (int m = 0; m < H; ++m) {
        const float qm = qv[m];
        mc1 += M1[m] * qm;
        mc2 += M2[m] * qm;
    }
    const float aS = 1.f / (1.f + __expf(-alpha_p[0]));
    mixedL[j1] = fmaxf(aS * re1 + (1.f - aS) * mc1, 0.f);
    mixedL[j2] = fmaxf(aS * re2 + (1.f - aS) * mc2, 0.f);
    __syncthreads();

    float o1 = bout[j1], o2 = bout[j2];
    for (int m = 0; m < H; ++m) {
        const float mm = mixedL[m];
        o1 += Wout[j1 * H + m] * mm;
        o2 += Wout[j2 * H + m] * mm;
    }
    out[b * VOC + j1] = o1;
    out[b * VOC + j2] = o2;
}

// ---------------------------------------------------------------------------
extern "C" void kernel_launch(void* const* d_in, const int* in_sizes, int n_in,
                              void* d_out, int out_size, void* d_ws, size_t ws_size,
                              hipStream_t stream) {
    const int*   x     = (const int*)  d_in[0];
    const float* embed = (const float*)d_in[1];
    const float* W1    = (const float*)d_in[2];
    const float* b1    = (const float*)d_in[3];
    const float* W2    = (const float*)d_in[4];
    const float* b2    = (const float*)d_in[5];
    const float* ln_g  = (const float*)d_in[6];
    const float* ln_b  = (const float*)d_in[7];
    const float* Wk    = (const float*)d_in[8];
    const float* Wv    = (const float*)d_in[9];
    const float* Wq    = (const float*)d_in[10];
    const float* Wr    = (const float*)d_in[11];
    const float* alpha = (const float*)d_in[12];
    const float* Wout  = (const float*)d_in[13];
    const float* bout  = (const float*)d_in[14];

    float* h_enc = (float*)d_ws;             // B*L*H f32 = 128 MB scratch
    float* out   = (float*)d_out;            // [B, VOCAB] f32

    enc_kernel<<<dim3(BSZ * LSQ / 32), dim3(64), 0, stream>>>(
        x, embed, W1, b1, W2, b2, ln_g, ln_b, h_enc);
    scan_kernel<<<dim3(BSZ), dim3(32), 0, stream>>>(
        h_enc, Wk, Wv, Wq, Wr, alpha, Wout, bout, out);
}
